// EquivariantPsuedoDecoder_1778116461237
// MI455X (gfx1250) — matrix operand order repaired and checked by reading, not compiled, not run on hardware
//
#include <hip/hip_runtime.h>
#include <hip/hip_bf16.h>
#include <math.h>

// ---------------- problem constants ----------------
#define NN      10000      // nodes
#define NE      160000     // edges
#define FD      128        // feature dim
#define NRBF    20
#define CUTOFF  5.0f
#define NCONV   3
#define EPSV    1e-15f

typedef __attribute__((ext_vector_type(16))) __bf16 v16bf;
typedef __attribute__((ext_vector_type(4)))  __bf16 v4bf;
typedef __attribute__((ext_vector_type(8)))  float  v8f;
typedef __attribute__((ext_vector_type(4)))  unsigned int v4u;

__device__ __forceinline__ float silu_f(float x) { return x / (1.0f + __expf(-x)); }

// ============================================================
// GEMM: C[M,N] = act(A[M,K] @ W[K,N] + bias[N])
// A,W fp32 row-major, converted to bf16 in LDS, f32 accumulate.
// block = 256 threads (8 waves), tile 128x64, K step 32.
// requires K % 32 == 0, N % 64 == 0 (true for all uses here).
// LDS layouts chosen so each lane's WMMA fragment is contiguous
// 16B-aligned chunks -> ds_load_b128, no b16 scatter/pack.
// ============================================================
template<bool ACT>
__global__ __launch_bounds__(256) void gemm_bf16_kernel(
    const float* __restrict__ A, const float* __restrict__ Wm,
    const float* __restrict__ bias, float* __restrict__ C,
    int M, int K, int N)
{
    // row stride 40 elems = 80 B -> 16B-aligned rows
    __shared__ __align__(16) __bf16 As[128][40];   // [m][k]
    __shared__ __align__(16) __bf16 Bt[64][40];    // [n][k]  (transposed tile)

    const int tid  = threadIdx.x;
    const int lane = tid & 31;
    const int wid  = tid >> 5;
    const int wrow = wid & 3;   // 0..3 -> M quadrant (32 rows each)
    const int wcol = wid >> 2;  // 0..1 -> N half (32 cols each)
    const int h    = lane >> 4; // lane half
    const int mn   = lane & 15;

    const int bm = blockIdx.x * 128;
    const int bn = blockIdx.y * 64;

    v8f acc[2][2];
    #pragma unroll
    for (int i = 0; i < 2; ++i)
        #pragma unroll
        for (int j = 0; j < 2; ++j)
            #pragma unroll
            for (int r = 0; r < 8; ++r)
                acc[i][j][r] = 0.0f;

    for (int k0 = 0; k0 < K; k0 += 32) {
        __syncthreads();
        // ---- stage A tile 128x32 : 1024 float4 / 256 thr = 4 each ----
        #pragma unroll
        for (int i = 0; i < 4; ++i) {
            int idx = i * 256 + tid;
            int r   = idx >> 3;          // 0..127
            int c4  = (idx & 7) * 4;     // 0,4,..,28
            int gr  = bm + r;
            int grc = (gr < M) ? gr : (M - 1);   // clamp, branchless
            const float* ap = A + (size_t)grc * K + (k0 + c4);
            float4 v = *(const float4*)ap;
            if (k0 + 32 < K) __builtin_prefetch(ap + 32, 0, 1);
            if (gr >= M) { v.x = 0.f; v.y = 0.f; v.z = 0.f; v.w = 0.f; }
            v4bf t = { (__bf16)v.x, (__bf16)v.y, (__bf16)v.z, (__bf16)v.w };
            *(v4bf*)&As[r][c4] = t;      // ds_store_b64
        }
        // ---- stage B tile 32x64 transposed : 512 float4 / 256 thr = 2 each ----
        #pragma unroll
        for (int i = 0; i < 2; ++i) {
            int idx = i * 256 + tid;
            int r   = idx >> 4;          // k row 0..31
            int c4  = (idx & 15) * 4;    // n col 0,4,..,60
            const float* wp = Wm + (size_t)(k0 + r) * N + (bn + c4);
            float4 v = *(const float4*)wp;
            if (k0 + 32 < K) __builtin_prefetch(wp + (size_t)32 * N, 0, 1);
            Bt[c4 + 0][r] = (__bf16)v.x;
            Bt[c4 + 1][r] = (__bf16)v.y;
            Bt[c4 + 2][r] = (__bf16)v.z;
            Bt[c4 + 3][r] = (__bf16)v.w;
        }
        __syncthreads();

        // ---- fragments ----
        // A (16-bit A 16x32 ISA layout): elem e -> k = e + 8h (e<8), e+8+8h (e>=8)
        v16bf aF[2];
        #pragma unroll
        for (int i = 0; i < 2; ++i) {
            int row = wrow * 32 + i * 16 + mn;
            ((v4u*)&aF[i])[0] = *(const v4u*)&As[row][8 * h];
            ((v4u*)&aF[i])[1] = *(const v4u*)&As[row][16 + 8 * h];
        }
        // B (16-bit B 32x16): elem e -> k = e + 8h (e<8), e+8+8h (e>=8)
        v16bf bF[2];
        #pragma unroll
        for (int j = 0; j < 2; ++j) {
            int col = wcol * 32 + j * 16 + mn;
            ((v4u*)&bF[j])[0] = *(const v4u*)&Bt[col][8 * h];
            ((v4u*)&bF[j])[1] = *(const v4u*)&Bt[col][16 + 8 * h];
        }
        #pragma unroll
        for (int i = 0; i < 2; ++i)
            #pragma unroll
            for (int j = 0; j < 2; ++j)
                acc[i][j] = __builtin_amdgcn_wmma_f32_16x16x32_bf16(
                    false, aF[i], false, bF[j], (short)0, acc[i][j], false, false);
    }

    // ---- store: C layout -> element r of v8f is (M = r + 8*half, N = mn) ----
    #pragma unroll
    for (int i = 0; i < 2; ++i) {
        #pragma unroll
        for (int j = 0; j < 2; ++j) {
            int gn = bn + wcol * 32 + j * 16 + mn;
            float bv = bias ? bias[gn] : 0.0f;
            #pragma unroll
            for (int r = 0; r < 8; ++r) {
                int gm = bm + wrow * 32 + i * 16 + r + 8 * h;
                if (gm < M) {
                    float val = acc[i][j][r] + bv;
                    if (ACT) val = silu_f(val);
                    C[(size_t)gm * N + gn] = val;
                }
            }
        }
    }
}

// ============================================================
// init: H <- Hin, Sbar/V/Vbar <- 0
// ============================================================
__global__ void init_state_kernel(const float* __restrict__ Hin,
                                  float* __restrict__ H, float* __restrict__ Sb,
                                  float* __restrict__ V, float* __restrict__ Vb)
{
    int idx = blockIdx.x * blockDim.x + threadIdx.x;
    if (idx < NN * FD * 3) { V[idx] = 0.0f; Vb[idx] = 0.0f; }
    if (idx < NN * FD)     { H[idx] = Hin[idx]; Sb[idx] = 0.0f; }
}

// ============================================================
// per-edge geometry: unit vector, RBF(20), envelope
// ============================================================
__global__ void edge_geom_kernel(const float* __restrict__ xyz,
                                 const int* __restrict__ nbr,
                                 float* __restrict__ unitv,
                                 float* __restrict__ rbfv,
                                 float* __restrict__ envv)
{
    int e = blockIdx.x * blockDim.x + threadIdx.x;
    if (e >= NE) return;
    int s = nbr[2 * e], d = nbr[2 * e + 1];
    float dx = xyz[3 * d + 0] - xyz[3 * s + 0];
    float dy = xyz[3 * d + 1] - xyz[3 * s + 1];
    float dz = xyz[3 * d + 2] - xyz[3 * s + 2];
    float dist = sqrtf(dx * dx + dy * dy + dz * dz);
    float inv = 1.0f / dist;
    unitv[3 * e + 0] = dx * inv;
    unitv[3 * e + 1] = dy * inv;
    unitv[3 * e + 2] = dz * inv;
    const float kpi = 3.14159265358979323846f / CUTOFF;
    #pragma unroll
    for (int r = 0; r < NRBF; ++r)
        rbfv[(size_t)e * NRBF + r] = sinf((float)(r + 1) * kpi * dist) * inv;
    envv[e] = (dist < CUTOFF) ? 0.5f * (cosf(kpi * dist) + 1.0f) : 0.0f;
}

// ============================================================
// edge messages: one block per edge, 128 threads (one per feature)
// w_s = (rbf @ dist_W + dist_b) * env ; inv = phi[dst] * w_s
// scatter: H[src]+=inv1, Sbar[src]+=inv3,
//          Vnew[src]+=inv2*unit + inv0*Vold[dst]
//          Vbnew[src]+=inv2*unit + inv4*Vbold[dst]
// ============================================================
__global__ __launch_bounds__(128) void edge_message_kernel(
    const float* __restrict__ phi, const float* __restrict__ rbfv,
    const float* __restrict__ envv, const float* __restrict__ unitv,
    const int* __restrict__ nbr,
    const float* __restrict__ dW, const float* __restrict__ db,
    const float* __restrict__ Vold, const float* __restrict__ Vbold,
    float* __restrict__ H, float* __restrict__ Sb,
    float* __restrict__ Vnew, float* __restrict__ Vbnew)
{
    int e = blockIdx.x;
    int f = threadIdx.x;
    __shared__ float s_rbf[NRBF];
    if (f < NRBF) s_rbf[f] = rbfv[(size_t)e * NRBF + f];
    __syncthreads();

    float ev = envv[e];
    int s = nbr[2 * e], d = nbr[2 * e + 1];
    float ux = unitv[3 * e + 0], uy = unitv[3 * e + 1], uz = unitv[3 * e + 2];

    float inv[5];
    #pragma unroll
    for (int c = 0; c < 5; ++c) {
        int col = c * FD + f;
        float acc = db[col];
        #pragma unroll
        for (int r = 0; r < NRBF; ++r)
            acc += s_rbf[r] * dW[r * (5 * FD) + col];
        inv[c] = phi[(size_t)d * (5 * FD) + col] * (acc * ev);
    }

    atomicAdd(&H[(size_t)s * FD + f],  inv[1]);
    atomicAdd(&Sb[(size_t)s * FD + f], inv[3]);

    size_t dbase = ((size_t)d * FD + f) * 3;
    size_t sbase = ((size_t)s * FD + f) * 3;
    float vdx = Vold[dbase + 0], vdy = Vold[dbase + 1], vdz = Vold[dbase + 2];
    atomicAdd(&Vnew[sbase + 0], inv[2] * ux + inv[0] * vdx);
    atomicAdd(&Vnew[sbase + 1], inv[2] * uy + inv[0] * vdy);
    atomicAdd(&Vnew[sbase + 2], inv[2] * uz + inv[0] * vdz);
    float wdx = Vbold[dbase + 0], wdy = Vbold[dbase + 1], wdz = Vbold[dbase + 2];
    atomicAdd(&Vbnew[sbase + 0], inv[2] * ux + inv[4] * wdx);
    atomicAdd(&Vbnew[sbase + 1], inv[2] * uy + inv[4] * wdy);
    atomicAdd(&Vbnew[sbase + 2], inv[2] * uz + inv[4] * wdz);
}

// ============================================================
// Vt[(n*3+d)*128 + g] = V[(n*128+g)*3 + d]  (rows = (n,d) pairs)
// ============================================================
__global__ void transpose_V_kernel(const float* __restrict__ V, float* __restrict__ Vt)
{
    int idx = blockIdx.x * blockDim.x + threadIdx.x;
    if (idx >= NN * 3 * FD) return;
    int g = idx & (FD - 1);
    int nd = idx >> 7;
    int d = nd % 3, n = nd / 3;
    Vt[idx] = V[((size_t)n * FD + g) * 3 + d];
}

// ============================================================
// v_norm[n,g] = sqrt(sum_d v_v[(n*3+d)*128+g]^2 + EPS)
// ============================================================
__global__ void vnorm_kernel(const float* __restrict__ vv, float* __restrict__ vn)
{
    int idx = blockIdx.x * blockDim.x + threadIdx.x;
    if (idx >= NN * FD) return;
    int g = idx & (FD - 1);
    int n = idx >> 7;
    float s = EPSV;
    #pragma unroll
    for (int d = 0; d < 3; ++d) {
        float v = vv[((size_t)n * 3 + d) * FD + g];
        s += v * v;
    }
    vn[idx] = sqrtf(s);
}

// ============================================================
// C[M,256] = [A | B], A,B each M x 128
// ============================================================
__global__ void concat2_kernel(const float* __restrict__ A, const float* __restrict__ B,
                               float* __restrict__ C, int M)
{
    int idx = blockIdx.x * blockDim.x + threadIdx.x;
    if (idx >= M * 2 * FD) return;
    int c = idx & (2 * FD - 1);
    int n = idx >> 8;
    C[idx] = (c < FD) ? A[(size_t)n * FD + c] : B[(size_t)n * FD + (c - FD)];
}

// ============================================================
// node update: V += u_v * a0 ; H += a1 * <u_v, v_v> + a2
// ============================================================
__global__ void node_update_kernel(const float* __restrict__ aO,
                                   const float* __restrict__ uv,
                                   const float* __restrict__ vv,
                                   float* __restrict__ H, float* __restrict__ V)
{
    int idx = blockIdx.x * blockDim.x + threadIdx.x;
    if (idx >= NN * FD) return;
    int g = idx & (FD - 1);
    int n = idx >> 7;
    float a0 = aO[(size_t)n * 3 * FD + g];
    float a1 = aO[(size_t)n * 3 * FD + FD + g];
    float a2 = aO[(size_t)n * 3 * FD + 2 * FD + g];
    float dot = 0.0f;
    #pragma unroll
    for (int d = 0; d < 3; ++d) {
        float u = uv[((size_t)n * 3 + d) * FD + g];
        float v = vv[((size_t)n * 3 + d) * FD + g];
        dot += u * v;
        V[((size_t)n * FD + g) * 3 + d] += u * a0;
    }
    H[idx] += a1 * dot + a2;
}

// ============================================================
// host launcher
// ============================================================
extern "C" void kernel_launch(void* const* d_in, const int* in_sizes, int n_in,
                              void* d_out, int out_size, void* d_ws, size_t ws_size,
                              hipStream_t stream)
{
    (void)in_sizes; (void)n_in; (void)out_size; (void)ws_size;

    const float* cg_xyz = (const float*)d_in[0];
    const int*   nbr    = (const int*)d_in[1];
    // d_in[2] = mapping (unused by reference)
    const float* Hin    = (const float*)d_in[3];
    const float* msg_W1 = (const float*)d_in[4];
    const float* msg_b1 = (const float*)d_in[5];
    const float* msg_W2 = (const float*)d_in[6];
    const float* msg_b2 = (const float*)d_in[7];
    const float* dist_W = (const float*)d_in[8];
    const float* dist_b = (const float*)d_in[9];
    const float* upd_U  = (const float*)d_in[10];
    const float* upd_V  = (const float*)d_in[11];
    const float* upd_W1 = (const float*)d_in[12];
    const float* upd_b1 = (const float*)d_in[13];
    const float* upd_W2 = (const float*)d_in[14];
    const float* upd_b2 = (const float*)d_in[15];

    // ---- workspace layout (floats) ----
    float* Wp = (float*)d_ws;
    size_t off = 0;
    float* unitv = Wp + off; off += (size_t)NE * 3;
    float* rbfv  = Wp + off; off += (size_t)NE * NRBF;
    float* envv  = Wp + off; off += (size_t)NE;
    float* Hs    = Wp + off; off += (size_t)NN * FD;
    float* Sbs   = Wp + off; off += (size_t)NN * FD;
    float* Vp0   = Wp + off; off += (size_t)NN * FD * 3;
    float* Vp1   = Wp + off; off += (size_t)NN * FD * 3;
    float* Vbp0  = Wp + off; off += (size_t)NN * FD * 3;
    float* Vbp1  = Wp + off; off += (size_t)NN * FD * 3;
    // overlapped scratch region (stage1 and stage2 lifetimes are disjoint)
    size_t sb = off;
    float* s_cat = Wp + sb;                       // N x 256
    float* X1    = s_cat + (size_t)NN * 256;      // N x 256
    float* phi   = X1 + (size_t)NN * 256;         // N x 640
    float* Vt    = Wp + sb;                       // 30000 x 128
    float* u_v   = Vt  + (size_t)NN * 3 * FD;     // 30000 x 128
    float* v_v   = u_v + (size_t)NN * 3 * FD;     // 30000 x 128
    float* vnrm  = v_v + (size_t)NN * 3 * FD;     // N x 128
    float* s_stk = vnrm + (size_t)NN * FD;        // N x 256
    float* X2    = s_stk + (size_t)NN * 256;      // N x 128
    float* aOut  = X2 + (size_t)NN * FD;          // N x 384

    const int TB = 256;
    dim3 blk(TB);
    int g_nf3 = (NN * FD * 3 + TB - 1) / TB;
    int g_nf  = (NN * FD + TB - 1) / TB;
    int g_n2f = (NN * 2 * FD + TB - 1) / TB;
    int g_e   = (NE + TB - 1) / TB;
    int gM_n  = (NN + 127) / 128;       // 79
    int gM_nv = (NN * 3 + 127) / 128;   // 235

    init_state_kernel<<<g_nf3, blk, 0, stream>>>(Hin, Hs, Sbs, Vp0, Vbp0);
    edge_geom_kernel<<<g_e, blk, 0, stream>>>(cg_xyz, nbr, unitv, rbfv, envv);

    float* Vcur = Vp0;  float* Vnxt = Vp1;
    float* Vbcur = Vbp0; float* Vbnxt = Vbp1;

    for (int l = 0; l < NCONV; ++l) {
        const float* mW1 = msg_W1 + (size_t)l * 256 * 256;
        const float* mb1 = msg_b1 + (size_t)l * 256;
        const float* mW2 = msg_W2 + (size_t)l * 256 * 640;
        const float* mb2 = msg_b2 + (size_t)l * 640;
        const float* dW  = dist_W + (size_t)l * NRBF * 640;
        const float* db  = dist_b + (size_t)l * 640;
        const float* uU  = upd_U  + (size_t)l * 128 * 128;
        const float* uV  = upd_V  + (size_t)l * 128 * 128;
        const float* uW1 = upd_W1 + (size_t)l * 256 * 128;
        const float* ub1 = upd_b1 + (size_t)l * 128;
        const float* uW2 = upd_W2 + (size_t)l * 128 * 384;
        const float* ub2 = upd_b2 + (size_t)l * 384;

        // phi = silu([H|Sbar] @ W1 + b1) @ W2 + b2
        concat2_kernel<<<g_n2f, blk, 0, stream>>>(Hs, Sbs, s_cat, NN);
        gemm_bf16_kernel<true ><<<dim3(gM_n, 4),  blk, 0, stream>>>(s_cat, mW1, mb1, X1,  NN, 256, 256);
        gemm_bf16_kernel<false><<<dim3(gM_n, 10), blk, 0, stream>>>(X1,    mW2, mb2, phi, NN, 256, 640);

        // double-buffer V / Vbar so edge kernel reads old, accumulates into new
        hipMemcpyAsync(Vnxt,  Vcur,  (size_t)NN * FD * 3 * sizeof(float), hipMemcpyDeviceToDevice, stream);
        hipMemcpyAsync(Vbnxt, Vbcur, (size_t)NN * FD * 3 * sizeof(float), hipMemcpyDeviceToDevice, stream);
        edge_message_kernel<<<dim3(NE), dim3(FD), 0, stream>>>(
            phi, rbfv, envv, unitv, nbr, dW, db, Vcur, Vbcur, Hs, Sbs, Vnxt, Vbnxt);
        { float* t = Vcur; Vcur = Vnxt; Vnxt = t; }
        { float* t = Vbcur; Vbcur = Vbnxt; Vbnxt = t; }

        // u_v / v_v via WMMA GEMMs on (n,d)-major rows
        transpose_V_kernel<<<g_nf3, blk, 0, stream>>>(Vcur, Vt);
        gemm_bf16_kernel<false><<<dim3(gM_nv, 2), blk, 0, stream>>>(Vt, uU, nullptr, u_v, NN * 3, 128, 128);
        gemm_bf16_kernel<false><<<dim3(gM_nv, 2), blk, 0, stream>>>(Vt, uV, nullptr, v_v, NN * 3, 128, 128);
        vnorm_kernel<<<g_nf, blk, 0, stream>>>(v_v, vnrm);

        // a = silu([H|v_norm] @ uW1 + ub1) @ uW2 + ub2
        concat2_kernel<<<g_n2f, blk, 0, stream>>>(Hs, vnrm, s_stk, NN);
        gemm_bf16_kernel<true ><<<dim3(gM_n, 2), blk, 0, stream>>>(s_stk, uW1, ub1, X2,   NN, 256, 128);
        gemm_bf16_kernel<false><<<dim3(gM_n, 6), blk, 0, stream>>>(X2,    uW2, ub2, aOut, NN, 128, 384);

        node_update_kernel<<<g_nf, blk, 0, stream>>>(aOut, u_v, v_v, Hs, Vcur);
    }

    // output = (H, V) flat
    float* outp = (float*)d_out;
    hipMemcpyAsync(outp, Hs, (size_t)NN * FD * sizeof(float), hipMemcpyDeviceToDevice, stream);
    hipMemcpyAsync(outp + (size_t)NN * FD, Vcur, (size_t)NN * FD * 3 * sizeof(float), hipMemcpyDeviceToDevice, stream);
}
